// CFIW_net_32607391711724
// MI455X (gfx1250) — matrix operand order repaired and checked by reading, not compiled, not run on hardware
//
#include <hip/hip_runtime.h>
#include <hip/hip_bf16.h>
#include <math.h>

// ---------------------------------------------------------------------------
// CDNA5 (gfx1250) implementation of the CFIW multimodal net forward pass.
// Heavy GEMMs use v_wmma_f32_16x16x32_bf16 (bf16 operands, f32 accum).
// Fast GEMM path: 64x128 tile, b128 global loads -> v_cvt_pk_bf16_f32 pack ->
// b128 LDS stores, 4 WMMAs per wave per K-step. Generic guarded path for K=24.
// ---------------------------------------------------------------------------

typedef __attribute__((ext_vector_type(16))) __bf16 v16bf;
typedef __attribute__((ext_vector_type(8)))  float  v8f;
typedef __attribute__((ext_vector_type(2)))  __bf16 v2bf;

namespace {
constexpr int kB      = 16;
constexpr int kL      = 196;    // 14x14 patches
constexpr int kD      = 384;    // d_model
constexpr int kBL     = kB * kL;
constexpr int kDI     = 768;    // d_inner
constexpr int kHeads  = 12;
constexpr int kHd     = 32;     // head dim
constexpr int kDS     = 32;     // d_state
constexpr int kDTR    = 24;     // dt_rank
constexpr int kNC     = 5;      // classes
constexpr int kDbcLd  = kDTR + 2 * kDS; // 88
}

static inline int ceil_div(int a, int b) { return (a + b - 1) / b; }

// ------------------------------ device helpers -----------------------------

__device__ __forceinline__ float geluf(float x) {
  return 0.5f * x * (1.0f + erff(x * 0.70710678118654752f));
}
__device__ __forceinline__ float softplusf(float x) {
  return (x > 20.f) ? x : log1pf(expf(x));
}
__device__ __forceinline__ float siluf(float x) {
  return x / (1.f + expf(-x));
}

// Packed f32 -> bf16 conversion. Prefer the hardware packed cvt; fall back to
// scalar __bf16 casts (clang emits native cvt where available).
__device__ __forceinline__ unsigned pack_bf2(float lo, float hi) {
#if __has_builtin(__builtin_amdgcn_cvt_pk_bf16_f32)
  v2bf r = __builtin_amdgcn_cvt_pk_bf16_f32(lo, hi);
  union { v2bf v; unsigned u; } c; c.v = r;
  return c.u;
#else
  union { __bf16 h[2]; unsigned u; } c;
  c.h[0] = (__bf16)lo; c.h[1] = (__bf16)hi;
  return c.u;
#endif
}
__device__ __forceinline__ unsigned short f2bf(float f) {
  union { __bf16 h; unsigned short u; } c;
  c.h = (__bf16)f;
  return c.u;
}

// --------------------- fast WMMA bf16 GEMM (64x128 tile) -------------------
// C[M,N] = act(A[M,K] * W[N,K]^T + bias). Requires: M%64==0, K%32==0, lda%4==0,
// K%4==0 (W row stride). N arbitrary (uniform per-row guard on B tile + cols).
// Block: 256 threads = 8 waves. wave: wm=id>>1 (16-row strip), wn=id&1
// (64-col strip -> 4 accumulators).

__global__ __launch_bounds__(256) void k_gemm_bf16_fast(
    const float* __restrict__ A, int lda,
    const float* __restrict__ W,              // [N,K] row-major
    const float* __restrict__ bias,           // may be null
    float* __restrict__ C, int ldc,
    int M, int N, int K, int act)             // act: 0 none, 1 gelu, 2 softplus
{
  __shared__ unsigned short sA[64 * 32];
  __shared__ unsigned short sB[128 * 32];

  const int tid  = threadIdx.x;
  const int lane = tid & 31;
  const int wave = tid >> 5;       // 0..7
  const int wm   = wave >> 1;      // 0..3 : 16-row strip
  const int wn   = wave & 1;       // 0..1 : 64-col strip
  const int m0   = blockIdx.x * 64;
  const int n0   = blockIdx.y * 128;

  const int hi  = lane >> 4;       // K-half select per ISA layout
  const int l16 = lane & 15;

  // tile-staging coordinates
  const int rowA = tid >> 2;             // 0..63
  const int segA = (tid & 3) * 8;        // 0,8,16,24
  const int rowB = tid >> 1;             // 0..127
  const int segB = (tid & 1) * 16;       // 0,16
  const int gnB  = n0 + rowB;

  union FC { v8f v; float f[8]; } c0, c1, c2, c3;
  c0.v = v8f{}; c1.v = v8f{}; c2.v = v8f{}; c3.v = v8f{};

  for (int k0 = 0; k0 < K; k0 += 32) {
    // ---- stage A tile [64][32] : 2x b128 load, 1x b128 LDS store ----
    {
      const float* src = A + (size_t)(m0 + rowA) * lda + k0 + segA;
      float4 a0 = *(const float4*)src;
      float4 a1 = *(const float4*)(src + 4);
      uint4 p;
      p.x = pack_bf2(a0.x, a0.y); p.y = pack_bf2(a0.z, a0.w);
      p.z = pack_bf2(a1.x, a1.y); p.w = pack_bf2(a1.z, a1.w);
      *(uint4*)&sA[rowA * 32 + segA] = p;
    }
    // ---- stage B tile [128][32] : 4x b128 load, 2x b128 LDS store ----
    {
      uint4 p0 = {0u, 0u, 0u, 0u}, p1 = {0u, 0u, 0u, 0u};
      if (gnB < N) {
        const float* src = W + (size_t)gnB * K + k0 + segB;
        float4 b0 = *(const float4*)src;
        float4 b1 = *(const float4*)(src + 4);
        float4 b2 = *(const float4*)(src + 8);
        float4 b3 = *(const float4*)(src + 12);
        p0.x = pack_bf2(b0.x, b0.y); p0.y = pack_bf2(b0.z, b0.w);
        p0.z = pack_bf2(b1.x, b1.y); p0.w = pack_bf2(b1.z, b1.w);
        p1.x = pack_bf2(b2.x, b2.y); p1.y = pack_bf2(b2.z, b2.w);
        p1.z = pack_bf2(b3.x, b3.y); p1.w = pack_bf2(b3.z, b3.w);
      }
      *(uint4*)&sB[rowB * 32 + segB]     = p0;
      *(uint4*)&sB[rowB * 32 + segB + 8] = p1;
    }
    __syncthreads();

    // ---- fragments per CDNA5 16-bit WMMA VGPR layout (ISA 7.12.2) ----
    union FR { v16bf v; unsigned u[8]; } fa, fb0, fb1, fb2, fb3;
    const int mr = wm * 16 + l16;
#pragma unroll
    for (int j = 0; j < 8; ++j) {
      // A: lanes0-15: V0-3 -> K0..7, V4-7 -> K16..23 ; lanes16-31: +8
      int kk = ((j & 3) * 2) + ((j >> 2) * 16) + hi * 8;
      fa.u[j] = *(const unsigned*)&sA[mr * 32 + kk];
    }
    const int nrb = wn * 64 + l16;
#pragma unroll
    for (int j = 0; j < 8; ++j) {
      // B: lanes0-15: V0-3 -> K0..7, V4-7 -> K16..23 ; lanes16-31: +8
      int kk = ((j & 3) * 2) + ((j >> 2) * 16) + hi * 8;
      fb0.u[j] = *(const unsigned*)&sB[(nrb)      * 32 + kk];
      fb1.u[j] = *(const unsigned*)&sB[(nrb + 16) * 32 + kk];
      fb2.u[j] = *(const unsigned*)&sB[(nrb + 32) * 32 + kk];
      fb3.u[j] = *(const unsigned*)&sB[(nrb + 48) * 32 + kk];
    }
    c0.v = __builtin_amdgcn_wmma_f32_16x16x32_bf16(false, fa.v, false, fb0.v,
                                                   (short)0, c0.v, false, false);
    c1.v = __builtin_amdgcn_wmma_f32_16x16x32_bf16(false, fa.v, false, fb1.v,
                                                   (short)0, c1.v, false, false);
    c2.v = __builtin_amdgcn_wmma_f32_16x16x32_bf16(false, fa.v, false, fb2.v,
                                                   (short)0, c2.v, false, false);
    c3.v = __builtin_amdgcn_wmma_f32_16x16x32_bf16(false, fa.v, false, fb3.v,
                                                   (short)0, c3.v, false, false);
    __syncthreads();
  }

  // ---- epilogue. C layout: lanes0-15: VGPR r -> (M=r,N=lane); 16-31: M=8+r.
  const float* frs[4] = { c0.f, c1.f, c2.f, c3.f };
#pragma unroll
  for (int sub = 0; sub < 4; ++sub) {
    int col = n0 + wn * 64 + sub * 16 + l16;
    if (col >= N) continue;
    float bv = bias ? bias[col] : 0.f;
    const float* fr = frs[sub];
#pragma unroll
    for (int r = 0; r < 8; ++r) {
      int gr = m0 + wm * 16 + hi * 8 + r;
      float v = fr[r] + bv;
      if (act == 1) v = geluf(v);
      else if (act == 2) v = softplusf(v);
      C[(size_t)gr * ldc + col] = v;
    }
  }
}

// --------------- generic guarded WMMA GEMM (only K=24 case) ----------------

__global__ __launch_bounds__(256) void k_gemm_bf16_gen(
    const float* __restrict__ A, int lda,
    const float* __restrict__ W,
    const float* __restrict__ bias,
    float* __restrict__ C, int ldc,
    int M, int N, int K, int act)
{
  __shared__ unsigned short sA[64 * 32];
  __shared__ unsigned short sB[64 * 32];

  const int tid  = threadIdx.x;
  const int lane = tid & 31;
  const int wave = tid >> 5;
  const int wm   = wave >> 1;
  const int wn   = wave & 1;
  const int m0   = blockIdx.x * 64;
  const int n0   = blockIdx.y * 64;

  const int hi  = lane >> 4;
  const int l16 = lane & 15;

  union FC { v8f v; float f[8]; } c0, c1;
  c0.v = v8f{}; c1.v = v8f{};

  for (int k0 = 0; k0 < K; k0 += 32) {
    for (int i = tid; i < 64 * 32; i += 256) {
      int r = i >> 5, cc = i & 31;
      int gm = m0 + r, gk = k0 + cc;
      float v = (gm < M && gk < K) ? A[(size_t)gm * lda + gk] : 0.0f;
      sA[i] = f2bf(v);
    }
    for (int i = tid; i < 64 * 32; i += 256) {
      int r = i >> 5, cc = i & 31;
      int gn = n0 + r, gk = k0 + cc;
      float v = (gn < N && gk < K) ? W[(size_t)gn * K + gk] : 0.0f;
      sB[i] = f2bf(v);
    }
    __syncthreads();

    union FR { v16bf v; unsigned u[8]; } fa, fb0, fb1;
    const int mr = wm * 16 + l16;
#pragma unroll
    for (int j = 0; j < 8; ++j) {
      int kk = ((j & 3) * 2) + ((j >> 2) * 16) + hi * 8;
      fa.u[j] = *(const unsigned*)&sA[mr * 32 + kk];
    }
    const int nr0 = wn * 32 + l16;
#pragma unroll
    for (int j = 0; j < 8; ++j) {
      int kk = ((j & 3) * 2) + ((j >> 2) * 16) + hi * 8;
      fb0.u[j] = *(const unsigned*)&sB[(nr0)      * 32 + kk];
      fb1.u[j] = *(const unsigned*)&sB[(nr0 + 16) * 32 + kk];
    }
    c0.v = __builtin_amdgcn_wmma_f32_16x16x32_bf16(false, fa.v, false, fb0.v,
                                                   (short)0, c0.v, false, false);
    c1.v = __builtin_amdgcn_wmma_f32_16x16x32_bf16(false, fa.v, false, fb1.v,
                                                   (short)0, c1.v, false, false);
    __syncthreads();
  }

#pragma unroll
  for (int sub = 0; sub < 2; ++sub) {
    const float* fr = sub ? c1.f : c0.f;
    int col = n0 + wn * 32 + sub * 16 + l16;
    if (col >= N) continue;
    float bv = bias ? bias[col] : 0.f;
#pragma unroll
    for (int r = 0; r < 8; ++r) {
      int gr = m0 + wm * 16 + hi * 8 + r;
      if (gr >= M) continue;
      float v = fr[r] + bv;
      if (act == 1) v = geluf(v);
      else if (act == 2) v = softplusf(v);
      C[(size_t)gr * ldc + col] = v;
    }
  }
}

// --------------------------- direct conv + BN (+res)(+gelu) ----------------

__global__ void k_conv_bn(const float* __restrict__ x, const float* __restrict__ w,
                          const float* __restrict__ cbias,
                          const float* __restrict__ bg, const float* __restrict__ bb,
                          const float* __restrict__ bm, const float* __restrict__ bv,
                          const float* __restrict__ res, float* __restrict__ out,
                          int Bn, int Ci, int Hi, int Wi, int Co, int Ho, int Wo,
                          int K, int stride, int pad, int dil, int act)
{
  int idx = blockIdx.x * blockDim.x + threadIdx.x;
  int total = Bn * Co * Ho * Wo;
  if (idx >= total) return;
  int wo = idx % Wo; int t = idx / Wo;
  int ho = t % Ho; t /= Ho;
  int co = t % Co; int b = t / Co;

  float acc = cbias[co];
  const float* wp = w + (size_t)co * Ci * K * K;
  for (int ci = 0; ci < Ci; ++ci) {
    const float* xp = x + ((size_t)(b * Ci + ci)) * Hi * Wi;
    for (int ky = 0; ky < K; ++ky) {
      int hy = ho * stride - pad + ky * dil;
      if ((unsigned)hy >= (unsigned)Hi) continue;
      for (int kx = 0; kx < K; ++kx) {
        int wx = wo * stride - pad + kx * dil;
        if ((unsigned)wx >= (unsigned)Wi) continue;
        acc += xp[hy * Wi + wx] * wp[(ci * K + ky) * K + kx];
      }
    }
  }
  float inv = rsqrtf(bv[co] + 1e-5f);
  acc = (acc - bm[co]) * (inv * bg[co]) + bb[co];
  if (res) acc += res[idx];
  if (act) acc = geluf(acc);
  out[idx] = acc;
}

// --------------------- NCHW(14x14) -> tokens + cls + pos -------------------

__global__ void k_tok(const float* __restrict__ x, const float* __restrict__ cls,
                      const float* __restrict__ pos, float* __restrict__ out,
                      int Bn, int C, int HW)
{
  int idx = blockIdx.x * blockDim.x + threadIdx.x;
  int total = Bn * HW * C;
  if (idx >= total) return;
  int c = idx % C; int t = idx / C;
  int l = t % HW;  int b = t / HW;
  out[idx] = x[((size_t)(b * C + c)) * HW + l] + cls[c] + pos[l * C + c];
}

// ------------------------------ layernorm ----------------------------------
// one wave32 per token

__global__ void k_layernorm(const float* __restrict__ x, const float* __restrict__ g,
                            const float* __restrict__ b, float* __restrict__ out,
                            int T, int Dd)
{
  int wv = (blockIdx.x * blockDim.x + threadIdx.x) >> 5;
  int lane = threadIdx.x & 31;
  if (wv >= T) return;
  const float* xp = x + (size_t)wv * Dd;
  float s = 0.f, s2 = 0.f;
  for (int i = lane; i < Dd; i += 32) { float v = xp[i]; s += v; s2 += v * v; }
#pragma unroll
  for (int o = 16; o > 0; o >>= 1) {
    s  += __shfl_xor(s,  o, 32);
    s2 += __shfl_xor(s2, o, 32);
  }
  float mu = s / Dd;
  float inv = rsqrtf(s2 / Dd - mu * mu + 1e-5f);
  float* op = out + (size_t)wv * Dd;
  for (int i = lane; i < Dd; i += 32) op[i] = (xp[i] - mu) * inv * g[i] + b[i];
}

// ---------------- mamba: causal depthwise conv(4) + SiLU -------------------

__global__ void k_dwconv_silu(const float* __restrict__ xz, const float* __restrict__ cw,
                              const float* __restrict__ cb, float* __restrict__ xs,
                              int Bn, int Ln, int Di)
{
  int idx = blockIdx.x * blockDim.x + threadIdx.x;
  int total = Bn * Ln * Di;
  if (idx >= total) return;
  int d = idx % Di; int t = idx / Di;
  int l = t % Ln;   int b = t / Ln;
  const float* col = xz + (size_t)b * Ln * (2 * Di) + d; // xi half of in_proj out
  float acc = cb[d];
#pragma unroll
  for (int j = 0; j < 4; ++j) {
    int ll = l - 3 + j;
    if (ll >= 0) acc += cw[d * 4 + j] * col[(size_t)ll * 2 * Di];
  }
  xs[idx] = siluf(acc);
}

// ------------------- mamba selective-scan (state in regs) ------------------

__global__ void k_scan(const float* __restrict__ dt, const float* __restrict__ dbc,
                       const float* __restrict__ xs, const float* __restrict__ xz,
                       const float* __restrict__ A_log, const float* __restrict__ Dp,
                       float* __restrict__ y, int Bn, int Ln, int Di, int dbc_ld)
{
  int idx = blockIdx.x * blockDim.x + threadIdx.x;
  if (idx >= Bn * Di) return;
  int d = idx % Di, b = idx / Di;
  float Av[kDS], h[kDS];
#pragma unroll
  for (int s = 0; s < kDS; ++s) { Av[s] = -expf(A_log[d * kDS + s]); h[s] = 0.f; }
  float Dd = Dp[d];
  for (int l = 0; l < Ln; ++l) {
    size_t tok = (size_t)(b * Ln + l);
    float dtv = dt[tok * Di + d];
    float xv  = xs[tok * Di + d];
    const float* Bm = dbc + tok * dbc_ld + kDTR;        // B: [24..56)
    const float* Cm = Bm + kDS;                         // C: [56..88)
    float dx = dtv * xv;
    float acc = 0.f;
#pragma unroll
    for (int s = 0; s < kDS; ++s) {
      h[s] = h[s] * expf(dtv * Av[s]) + dx * Bm[s];
      acc += h[s] * Cm[s];
    }
    float yv = acc + xv * Dd;
    float zv = xz[tok * 2 * Di + Di + d];
    y[tok * Di + d] = yv * siluf(zv);
  }
}

// ---------------------- cross-attention (per-query thread) -----------------
// q: [B,L,384] (head h at h*32); kv: [B,L,768] with k at +h*32, v at +384+h*32.
// Writes out[b,l, out_off + h*32 + c] with row stride out_ld.

__global__ void k_attn(const float* __restrict__ q, const float* __restrict__ kv,
                       float* __restrict__ out, int out_ld, int out_off,
                       int Bn, int Ln, int do_clamp)
{
  int idx = blockIdx.x * blockDim.x + threadIdx.x;
  int total = Bn * kHeads * Ln;
  if (idx >= total) return;
  int qi = idx % Ln; int t = idx / Ln;
  int h = t % kHeads; int b = t / kHeads;

  const float* qp = q + ((size_t)(b * Ln + qi)) * kD + h * kHd;
  const float* kb = kv + (size_t)b * Ln * (2 * kD) + h * kHd;
  const float* vb = kb + kD;
  const float scale = 0.17677669529663687f; // 32^-0.5

  float mx = -1e30f;
  for (int j = 0; j < Ln; ++j) {
    const float* kp = kb + (size_t)j * (2 * kD);
    float dot = 0.f;
#pragma unroll
    for (int c = 0; c < kHd; ++c) dot += qp[c] * kp[c];
    dot *= scale;
    if (do_clamp) dot = fminf(fmaxf(dot, -100.f), 100.f);
    mx = fmaxf(mx, dot);
  }
  float o[kHd];
#pragma unroll
  for (int c = 0; c < kHd; ++c) o[c] = 0.f;
  float sum = 0.f;
  for (int j = 0; j < Ln; ++j) {
    const float* kp = kb + (size_t)j * (2 * kD);
    float dot = 0.f;
#pragma unroll
    for (int c = 0; c < kHd; ++c) dot += qp[c] * kp[c];
    dot *= scale;
    if (do_clamp) dot = fminf(fmaxf(dot, -100.f), 100.f);
    float e = expf(dot - mx);
    sum += e;
    const float* vp = vb + (size_t)j * (2 * kD);
#pragma unroll
    for (int c = 0; c < kHd; ++c) o[c] += e * vp[c];
  }
  float inv = 1.f / sum;
  float* op = out + ((size_t)(b * Ln + qi)) * out_ld + out_off + h * kHd;
#pragma unroll
  for (int c = 0; c < kHd; ++c) op[c] = o[c] * inv;
}

// ------------------------------- misc --------------------------------------

__global__ void k_add3(float* __restrict__ x, const float* __restrict__ a,
                       const float* __restrict__ b, int n)
{
  int i = blockIdx.x * blockDim.x + threadIdx.x;
  if (i < n) x[i] += a[i] + b[i];
}

__global__ void k_maxpool(const float* __restrict__ x, float* __restrict__ out,
                          int Bn, int Ln, int Dd)
{
  int i = blockIdx.x * blockDim.x + threadIdx.x;
  if (i >= Bn * Dd) return;
  int d = i % Dd, b = i / Dd;
  float m = -1e30f;
  for (int l = 0; l < Ln; ++l) m = fmaxf(m, x[((size_t)(b * Ln + l)) * Dd + d]);
  out[i] = m;
}

__global__ void k_head(const float* __restrict__ x, const float* __restrict__ w,
                       const float* __restrict__ b, float* __restrict__ out,
                       int Bn, int Dd, int Nc)
{
  int i = blockIdx.x * blockDim.x + threadIdx.x;
  if (i >= Bn * Nc) return;
  int n = i % Nc, bb = i / Nc;
  float acc = b[n];
  for (int d = 0; d < Dd; ++d) acc += x[(size_t)bb * Dd + d] * w[(size_t)n * Dd + d];
  out[i] = acc;
}

// ---------------------------------------------------------------------------
// Host side: parameter cursor (JAX pytree / alphabetical-dict-key order),
// workspace layout, and orchestration.
// ---------------------------------------------------------------------------

namespace {

struct LinP  { const float *b, *w; };
struct LnP   { const float *b, *g; };
struct BnP   { const float *b, *g, *m, *v; };
struct ConvP { const float *b, *w; };
struct MambaP{ const float *A_log, *D, *cb, *cw, *dt_b, *dt_w, *in_w, *out_w, *xp_w; };
struct RmP   { LinP fc1, fc2; LnP ln1, ln2; MambaP m; };
struct TcnP  { BnP bn1, bn2; ConvP c1, c2; BnP sbn; ConvP sc; };
struct StemP { RmP blk; const float* cls; LnP ln; const float* pos; BnP sm_bn; ConvP sm_c; TcnP tcn[4]; };
struct FusP  { LinP kv_i, kv_w; LnP ln; LinP proj, q_i, q_w; };

struct Cur {
  void* const* d; int i;
  const float* n() { return (const float*)d[i++]; }
};
static LinP  rdLin (Cur& c){ LinP p;  p.b=c.n(); p.w=c.n(); return p; }
static LnP   rdLn  (Cur& c){ LnP p;   p.b=c.n(); p.g=c.n(); return p; }
static BnP   rdBn  (Cur& c){ BnP p;   p.b=c.n(); p.g=c.n(); p.m=c.n(); p.v=c.n(); return p; }
static ConvP rdConv(Cur& c){ ConvP p; p.b=c.n(); p.w=c.n(); return p; }
static MambaP rdM  (Cur& c){ MambaP m; m.A_log=c.n(); m.D=c.n(); m.cb=c.n(); m.cw=c.n();
                             m.dt_b=c.n(); m.dt_w=c.n(); m.in_w=c.n(); m.out_w=c.n();
                             m.xp_w=c.n(); return m; }
static RmP   rdRm  (Cur& c){ RmP r; r.fc1=rdLin(c); r.fc2=rdLin(c); r.ln1=rdLn(c);
                             r.ln2=rdLn(c); r.m=rdM(c); return r; }
static TcnP  rdTcn (Cur& c){ TcnP t; t.bn1=rdBn(c); t.bn2=rdBn(c); t.c1=rdConv(c);
                             t.c2=rdConv(c); t.sbn=rdBn(c); t.sc=rdConv(c); return t; }
static StemP rdStem(Cur& c){ StemP s; s.blk=rdRm(c); s.cls=c.n(); s.ln=rdLn(c);
                             s.pos=c.n(); s.sm_bn=rdBn(c); s.sm_c=rdConv(c);
                             for (int i=0;i<4;++i) s.tcn[i]=rdTcn(c); return s; }
static FusP  rdFus (Cur& c){ FusP f; f.kv_i=rdLin(c); f.kv_w=rdLin(c); f.ln=rdLn(c);
                             f.proj=rdLin(c); f.q_i=rdLin(c); f.q_w=rdLin(c); return f; }

struct Ws {
  float *W0, *W1, *W2, *W3;               // conv ping-pong (max 16*48*112*112)
  float *tok_w, *tok_s, *tok_f, *tok_x;   // [B,L,384]
  float *t1, *t2, *t4, *m_out;            // [B,L,384]
  float *t3;                              // [B,L,1536]
  float *xz;                              // [B,L,1536]
  float *xs, *dtb, *yact;                 // [B,L,768]
  float *dbc;                             // [B,L,88]
  float *fq;                              // [B,L,384]
  float *fkv, *concat;                    // [B,L,768]
  float *pooled;                          // [B,384]
};

struct Alloc {
  char* p; size_t off;
  float* get(size_t nfloats) {
    float* r = (float*)(p + off);
    off += ((nfloats * sizeof(float) + 255) / 256) * 256;
    return r;
  }
};

static void gemm(const float* A, int lda, const float* W, const float* bias,
                 float* C, int ldc, int M, int N, int K, int act, hipStream_t s)
{
  if ((M % 64) == 0 && (K % 32) == 0 && (lda % 4) == 0) {
    dim3 g(M / 64, ceil_div(N, 128));
    k_gemm_bf16_fast<<<g, 256, 0, s>>>(A, lda, W, bias, C, ldc, M, N, K, act);
  } else {
    dim3 g(ceil_div(M, 64), ceil_div(N, 64));
    k_gemm_bf16_gen<<<g, 256, 0, s>>>(A, lda, W, bias, C, ldc, M, N, K, act);
  }
}

static void conv(const float* x, const ConvP& cp, const BnP& bn, const float* res,
                 float* out, int Bn, int Ci, int Hi, int Wi, int Co, int Ho, int Wo,
                 int K, int stride, int pad, int dil, int act, hipStream_t s)
{
  int total = Bn * Co * Ho * Wo;
  k_conv_bn<<<ceil_div(total, 256), 256, 0, s>>>(
      x, cp.w, cp.b, bn.g, bn.b, bn.m, bn.v, res, out,
      Bn, Ci, Hi, Wi, Co, Ho, Wo, K, stride, pad, dil, act);
}

static void layernorm(const float* x, const LnP& p, float* out, hipStream_t s)
{
  int waves = kBL;                    // one wave per token
  k_layernorm<<<ceil_div(waves * 32, 256), 256, 0, s>>>(x, p.g, p.b, out, kBL, kD);
}

static void run_mamba(const float* xin, const MambaP& mp, Ws& w, hipStream_t s)
{
  gemm(xin, kD, mp.in_w, nullptr, w.xz, 2 * kDI, kBL, 2 * kDI, kD, 0, s);
  {
    int total = kBL * kDI;
    k_dwconv_silu<<<ceil_div(total, 256), 256, 0, s>>>(w.xz, mp.cw, mp.cb, w.xs,
                                                       kB, kL, kDI);
  }
  gemm(w.xs, kDI, mp.xp_w, nullptr, w.dbc, kDbcLd, kBL, kDbcLd, kDI, 0, s);
  gemm(w.dbc, kDbcLd, mp.dt_w, mp.dt_b, w.dtb, kDI, kBL, kDI, kDTR, 2 /*softplus*/, s);
  {
    int total = kB * kDI;
    k_scan<<<ceil_div(total, 256), 256, 0, s>>>(w.dtb, w.dbc, w.xs, w.xz,
                                                mp.A_log, mp.D, w.yact,
                                                kB, kL, kDI, kDbcLd);
  }
  gemm(w.yact, kDI, mp.out_w, nullptr, w.m_out, kD, kBL, kD, kDI, 0, s);
}

static void run_rm(float* x, const RmP& rp, Ws& w, hipStream_t s)
{
  layernorm(x, rp.ln1, w.t1, s);
  run_mamba(w.t1, rp.m, w, s);                              // -> w.m_out
  layernorm(x, rp.ln2, w.t2, s);
  gemm(w.t2, kD, rp.fc1.w, rp.fc1.b, w.t3, 4 * kD, kBL, 4 * kD, kD, 1 /*gelu*/, s);
  gemm(w.t3, 4 * kD, rp.fc2.w, rp.fc2.b, w.t4, kD, kBL, kD, 4 * kD, 0, s);
  k_add3<<<ceil_div(kBL * kD, 256), 256, 0, s>>>(x, w.m_out, w.t4, kBL * kD);
}

static void run_stem(const float* x_in, int Ci0, const StemP& sp, Ws& w,
                     float* out_tok, hipStream_t s)
{
  static const int ch[4] = {48, 96, 192, 384};
  static const int Hs[5] = {224, 112, 56, 28, 14};
  const float* cur = x_in;
  int ci = Ci0;
  float* outb = nullptr;
  for (int i = 0; i < 4; ++i) {
    int Hi = Hs[i], Ho = Hs[i + 1], co = ch[i], dil = 1 << i;
    outb = (i & 1) ? w.W3 : w.W2;
    // shortcut: 1x1 stride2 + bn
    conv(cur, sp.tcn[i].sc, sp.tcn[i].sbn, nullptr, w.W0,
         kB, ci, Hi, Hi, co, Ho, Ho, 1, 2, 0, 1, 0, s);
    // c1: 3x3 stride2 pad=dil dil + bn + gelu
    conv(cur, sp.tcn[i].c1, sp.tcn[i].bn1, nullptr, w.W1,
         kB, ci, Hi, Hi, co, Ho, Ho, 3, 2, dil, dil, 1, s);
    // c2: 3x3 stride1 pad=dil dil + bn + idt + gelu
    conv(w.W1, sp.tcn[i].c2, sp.tcn[i].bn2, w.W0, outb,
         kB, co, Ho, Ho, co, Ho, Ho, 3, 1, dil, dil, 1, s);
    cur = outb; ci = co;
  }
  // sm conv: 3x3 s1 p1 + bn + gelu at 14x14, 384->384
  float* smo = (cur == w.W3) ? w.W2 : w.W3;
  conv(cur, sp.sm_c, sp.sm_bn, nullptr, smo,
       kB, kD, 14, 14, kD, 14, 14, 3, 1, 1, 1, 1, s);
  // tokens + cls + pos
  k_tok<<<ceil_div(kBL * kD, 256), 256, 0, s>>>(smo, sp.cls, sp.pos, w.tok_w,
                                                kB, kD, kL);
  run_rm(w.tok_w, sp.blk, w, s);
  layernorm(w.tok_w, sp.ln, out_tok, s);
}

static void run_fusion(const float* tok_s, const float* tok_f, const FusP& fp,
                       Ws& w, float* out, hipStream_t s)
{
  int nq = kB * kHeads * kL;
  // s -> f attention (clamped scores)
  gemm(tok_s, kD, fp.q_i.w, fp.q_i.b, w.fq, kD, kBL, kD, kD, 0, s);
  gemm(tok_f, kD, fp.kv_w.w, fp.kv_w.b, w.fkv, 2 * kD, kBL, 2 * kD, kD, 0, s);
  k_attn<<<ceil_div(nq, 256), 256, 0, s>>>(w.fq, w.fkv, w.concat, 2 * kD, 0,
                                           kB, kL, 1);
  // f -> s attention
  gemm(tok_f, kD, fp.q_w.w, fp.q_w.b, w.fq, kD, kBL, kD, kD, 0, s);
  gemm(tok_s, kD, fp.kv_i.w, fp.kv_i.b, w.fkv, 2 * kD, kBL, 2 * kD, kD, 0, s);
  k_attn<<<ceil_div(nq, 256), 256, 0, s>>>(w.fq, w.fkv, w.concat, 2 * kD, kD,
                                           kB, kL, 0);
  // proj + ln
  gemm(w.concat, 2 * kD, fp.proj.w, fp.proj.b, w.t1, kD, kBL, kD, 2 * kD, 0, s);
  layernorm(w.t1, fp.ln, out, s);
}

} // namespace

// ---------------------------------------------------------------------------

extern "C" void kernel_launch(void* const* d_in, const int* in_sizes, int n_in,
                              void* d_out, int out_size, void* d_ws, size_t ws_size,
                              hipStream_t stream)
{
  (void)in_sizes; (void)n_in; (void)out_size; (void)ws_size;

  const float* x_img  = (const float*)d_in[0];  // [16,3,224,224]
  const float* x_wave = (const float*)d_in[1];  // [16,15,224,224]

  // Parameters in JAX pytree (alphabetical dict key) flatten order.
  Cur c{d_in, 2};
  FusP fus   = rdFus(c);
  LinP head  = rdLin(c);
  StemP img  = rdStem(c);
  LnP norm   = rdLn(c);
  RmP post[4];
  for (int i = 0; i < 4; ++i) post[i] = rdRm(c);
  StemP wav  = rdStem(c);

  // Workspace layout
  Alloc al{(char*)d_ws, 0};
  Ws w;
  const size_t bigN = (size_t)kB * 48 * 112 * 112;   // largest conv map
  w.W0     = al.get(bigN);
  w.W1     = al.get(bigN);
  w.W2     = al.get(bigN);
  w.W3     = al.get(bigN);
  w.tok_w  = al.get((size_t)kBL * kD);
  w.tok_s  = al.get((size_t)kBL * kD);
  w.tok_f  = al.get((size_t)kBL * kD);
  w.tok_x  = al.get((size_t)kBL * kD);
  w.t1     = al.get((size_t)kBL * kD);
  w.t2     = al.get((size_t)kBL * kD);
  w.t4     = al.get((size_t)kBL * kD);
  w.m_out  = al.get((size_t)kBL * kD);
  w.t3     = al.get((size_t)kBL * 4 * kD);
  w.xz     = al.get((size_t)kBL * 2 * kDI);
  w.xs     = al.get((size_t)kBL * kDI);
  w.dtb    = al.get((size_t)kBL * kDI);
  w.yact   = al.get((size_t)kBL * kDI);
  w.dbc    = al.get((size_t)kBL * kDbcLd);
  w.fq     = al.get((size_t)kBL * kD);
  w.fkv    = al.get((size_t)kBL * 2 * kD);
  w.concat = al.get((size_t)kBL * 2 * kD);
  w.pooled = al.get((size_t)kB * kD);

  // -------- forward pass --------
  run_stem(x_img, 3,  img, w, w.tok_s, stream);
  run_stem(x_wave, 15, wav, w, w.tok_f, stream);
  run_fusion(w.tok_s, w.tok_f, fus, w, w.tok_x, stream);
  for (int i = 0; i < 4; ++i) run_rm(w.tok_x, post[i], w, stream);

  layernorm(w.tok_x, norm, w.t1, stream);
  k_maxpool<<<ceil_div(kB * kD, 256), 256, 0, stream>>>(w.t1, w.pooled, kB, kL, kD);
  k_head<<<ceil_div(kB * kNC, 256), 256, 0, stream>>>(w.pooled, head.w, head.b,
                                                      (float*)d_out, kB, kD, kNC);
}
